// MSDSA_77266461655531
// MI455X (gfx1250) — compile-verified, matrix operand order checked by reading, not run on hardware
//
#include <hip/hip_runtime.h>
#include <hip/hip_bf16.h>

// ---------------------------------------------------------------------------
// MSDSA for MI455X (gfx1250): f16 WMMA GEMMs + flash-attention streaming
// softmax with Manhattan-decay bias. wave32, v_wmma_f32_16x16x32_f16.
//  - Q/K/V GEMM B operands stored transposed (n,c) -> all WMMA operand loads
//    are two contiguous 16B loads per lane.
//  - Softmax denominator accumulated by WMMA against an all-ones tile
//    (no cross-lane sum reduction; division is lane-local).
//  - Decay bias via fast __expf (single v_exp_f32, no precise-path fixup).
//  - attention kernel: __launch_bounds__(32,1) -> full VGPR budget, no spills.
// ---------------------------------------------------------------------------

typedef __attribute__((ext_vector_type(16))) _Float16 v16h;
typedef __attribute__((ext_vector_type(8)))  _Float16 v8h;
typedef __attribute__((ext_vector_type(8)))  float    v8f;

#define NPIX 4096   // 64*64
#define CDIM 256
#define NHEAD 8
#define HDIM 32

__device__ __forceinline__ v16h ld16h(const _Float16* p) {
  // 16 halves a lane owns for a 16x32 f16 A tile (or 32x16 B tile):
  // halves 0..7  -> K = koff+0..koff+7       (contiguous at p)
  // halves 8..15 -> K = 16+koff+0..16+koff+7 (contiguous at p+16)
  v8h lo = *(const v8h*)(p);
  v8h hi = *(const v8h*)(p + 16);
  v16h r;
#pragma unroll
  for (int i = 0; i < 8; ++i) { r[i] = lo[i]; r[i + 8] = hi[i]; }
  return r;
}

__device__ __forceinline__ v8f wmma_f16(v16h a, v16h b, v8f c) {
  return __builtin_amdgcn_wmma_f32_16x16x32_f16(
      false, a, false, b, (short)0, c, false, false);
}

// ---------------------------------------------------------------------------
// Combine the 3 depthwise kernels into one 7x7 per-channel kernel.
// ---------------------------------------------------------------------------
__global__ void build_wcomb_kernel(const float* __restrict__ w3,
                                   const float* __restrict__ w5,
                                   const float* __restrict__ w7,
                                   const float* __restrict__ ow,
                                   float* __restrict__ wcomb) {
  int t = blockIdx.x * blockDim.x + threadIdx.x;
  if (t >= CDIM * 49) return;
  int c = t / 49, k = t % 49;
  int ky = k / 7 - 3, kx = k % 7 - 3;
  float v = ow[2] * w7[c * 49 + k];
  if (ky >= -2 && ky <= 2 && kx >= -2 && kx <= 2)
    v += ow[1] * w5[c * 25 + (ky + 2) * 5 + (kx + 2)];
  if (ky >= -1 && ky <= 1 && kx >= -1 && kx <= 1)
    v += ow[0] * w3[c * 9 + (ky + 1) * 3 + (kx + 1)];
  wcomb[t] = v;
}

__global__ void cast_f16_kernel(const float* __restrict__ in,
                                _Float16* __restrict__ out, int n) {
  int t = blockIdx.x * blockDim.x + threadIdx.x;
  if (t < n) out[t] = (_Float16)in[t];
}

// f32 (c,n) -> f16 transposed (n,c)
__global__ void cast_f16_T_kernel(const float* __restrict__ in,
                                  _Float16* __restrict__ outT) {
  int t = blockIdx.x * 256 + threadIdx.x;   // CDIM*NPIX threads
  int c = t >> 12, p = t & 4095;
  outT[p * CDIM + c] = (_Float16)in[t];
}

// ---------------------------------------------------------------------------
// Fused depthwise conv (combined 7x7, SAME zero padding). feat[c*N + p].
// ---------------------------------------------------------------------------
__global__ void dwconv_kernel(const float* __restrict__ x,
                              const float* __restrict__ wcomb,
                              float* __restrict__ feat) {
  int t = blockIdx.x * 256 + threadIdx.x;   // CDIM*NPIX threads
  int c = t >> 12, p = t & 4095;
  int y = p >> 6, xcol = p & 63;
  const float* xc = x + c * NPIX;
  const float* wc = wcomb + c * 49;
  float acc = 0.f;
#pragma unroll
  for (int dy = -3; dy <= 3; ++dy) {
    int yy = y + dy;
    if (yy < 0 || yy > 63) continue;
#pragma unroll
    for (int dx = -3; dx <= 3; ++dx) {
      int xx = xcol + dx;
      if (xx < 0 || xx > 63) continue;
      acc += xc[yy * 64 + xx] * wc[(dy + 3) * 7 + (dx + 3)];
    }
  }
  feat[t] = acc;
}

// ---------------------------------------------------------------------------
// offset = w_offproj @ relu(feat) ; tanh * OFFSET_RANGE ; clip -> (gy,gx).
// ---------------------------------------------------------------------------
__global__ void offset_kernel(const float* __restrict__ feat,
                              const float* __restrict__ wop,
                              float* __restrict__ gxy) {
  __shared__ float s0[256];
  __shared__ float s1[256];
  int p = blockIdx.x, c = threadIdx.x;
  float f = fmaxf(feat[c * NPIX + p], 0.f);
  s0[c] = f * wop[c];           // row 0 -> y offset
  s1[c] = f * wop[CDIM + c];    // row 1 -> x offset
  __syncthreads();
  for (int st = 128; st > 0; st >>= 1) {
    if (c < st) { s0[c] += s0[c + st]; s1[c] += s1[c + st]; }
    __syncthreads();
  }
  if (c == 0) {
    int y = p >> 6, xcol = p & 63;
    float refy = ((y + 0.5f) / 63.f) * 2.f - 1.f;
    float refx = ((xcol + 0.5f) / 63.f) * 2.f - 1.f;
    float gy = refy + tanhf(s0[0]) * 2.f;
    float gx = refx + tanhf(s1[0]) * 2.f;
    gxy[p]        = fminf(fmaxf(gy, -1.f), 1.f);
    gxy[NPIX + p] = fminf(fmaxf(gx, -1.f), 1.f);
  }
}

// ---------------------------------------------------------------------------
// Bilinear grid sample (align_corners=True) -> f16 transposed (n,c) layout.
// ---------------------------------------------------------------------------
__global__ void gridsample_kernel(const float* __restrict__ x,
                                  const float* __restrict__ gxy,
                                  _Float16* __restrict__ xsT) {
  int t = blockIdx.x * 256 + threadIdx.x;
  int c = t >> 12, p = t & 4095;
  float gy = gxy[p], gx = gxy[NPIX + p];
  float ix = (gx + 1.f) * 0.5f * 63.f;
  float iy = (gy + 1.f) * 0.5f * 63.f;
  float x0f = floorf(ix), y0f = floorf(iy);
  float fx = ix - x0f, fy = iy - y0f;
  int x0 = min(max((int)x0f, 0), 63); int x1 = min(x0 + 1, 63);
  int y0 = min(max((int)y0f, 0), 63); int y1 = min(y0 + 1, 63);
  const float* xc = x + c * NPIX;
  float v00 = xc[y0 * 64 + x0], v01 = xc[y0 * 64 + x1];
  float v10 = xc[y1 * 64 + x0], v11 = xc[y1 * 64 + x1];
  float o = v00 * (1.f - fx) * (1.f - fy) + v01 * fx * (1.f - fy)
          + v10 * (1.f - fx) * fy + v11 * fx * fy;
  xsT[p * CDIM + c] = (_Float16)o;
}

// ---------------------------------------------------------------------------
// WMMA GEMM: out(256 x 4096) = W(256x256 row-major f16) x B(256x4096 f16).
// BLAYOUT 1: B given transposed (n,c) -> contiguous 16B tile loads.
// BLAYOUT 0: B given (c,n) -> strided element loads (final projection only).
// MODE 0: f16 out (c,n). MODE 1: f16 out transposed (n,c). MODE 2: f32 (c,n).
// ---------------------------------------------------------------------------
template <int MODE, int BLAYOUT>
__global__ void gemm256_kernel(const _Float16* __restrict__ W,
                               const _Float16* __restrict__ B,
                               void* __restrict__ out) {
  int lane = threadIdx.x & 31;
  int wv   = threadIdx.x >> 5;
  int m0   = blockIdx.x * 16;
  int n0   = (blockIdx.y * 8 + wv) * 16;
  int row  = lane & 15;
  int koff = (lane >> 4) * 8;
  int hi8  = koff;                // 8 * (lane>=16)
  int n    = n0 + row;
  v8f acc = {};
#pragma unroll
  for (int kb = 0; kb < 8; ++kb) {
    int k0 = kb * 32;
    v16h a = ld16h(W + (m0 + row) * CDIM + k0 + koff);
    v16h b;
    if (BLAYOUT == 1) {
      b = ld16h(B + n * CDIM + k0 + koff);
    } else {
#pragma unroll
      for (int i = 0; i < 8; ++i) {
        b[i]     = B[(k0 + koff + i) * NPIX + n];
        b[i + 8] = B[(k0 + 16 + koff + i) * NPIX + n];
      }
    }
    acc = wmma_f16(a, b, acc);
  }
  if (MODE == 1) {
    v8h o;
#pragma unroll
    for (int r = 0; r < 8; ++r) o[r] = (_Float16)acc[r];
    *(v8h*)((_Float16*)out + n * CDIM + m0 + hi8) = o;
  } else if (MODE == 0) {
#pragma unroll
    for (int r = 0; r < 8; ++r)
      ((_Float16*)out)[(m0 + r + hi8) * NPIX + n] = (_Float16)acc[r];
  } else {
#pragma unroll
    for (int r = 0; r < 8; ++r)
      ((float*)out)[(m0 + r + hi8) * NPIX + n] = acc[r];
  }
}

// ---------------------------------------------------------------------------
// Flash attention with Manhattan decay bias. One wave handles TWO 16-row Q
// tiles of one head (32 rows), streaming 32 keys/iter:
//   4x v_wmma (Q^T K) + online max + LDS transpose of P
//   + 4x v_wmma (P V) + 2x v_wmma (P * ones -> running denominator).
// Denominator lives in a WMMA accumulator: every column of that D tile holds
// the row-sum, so the final division is lane-local (no shuffle).
// __launch_bounds__(32,1): one wave per workgroup, full VGPR budget (the
// working set is ~180 VGPRs; default occupancy target caused scratch spills).
// qT,kT: (n,c) f16;  vm: (c,n) f16;  out attn_h: (c,n) f16.
// ---------------------------------------------------------------------------
__global__ void __launch_bounds__(32, 1)
attention_kernel(const _Float16* __restrict__ qT,
                 const _Float16* __restrict__ kT,
                 const _Float16* __restrict__ vm,
                 const float* __restrict__ beta_p,
                 _Float16* __restrict__ attn_h) {
  __shared__ _Float16 pbuf[2 * 16 * 32];
  int lane = threadIdx.x;
  float bden = fabsf(beta_p[0]) + 1e-6f;
  float c1 = -1.f / bden;   // decay = __expf(dist * c1)

  int h  = blockIdx.y;
  int n0 = blockIdx.x * 32;           // two 16-row tiles: n0, n0+16
  int row  = lane & 15;
  int hi8  = (lane >> 4) * 8;
  int koff = hi8;

  v16h aq0 = ld16h(qT + (n0 + row) * CDIM + h * HDIM + koff);
  v16h aq1 = ld16h(qT + (n0 + 16 + row) * CDIM + h * HDIM + koff);

  v16h vone;
#pragma unroll
  for (int i = 0; i < 16; ++i) vone[i] = (_Float16)1.f;

  float mr0[8], mr1[8];
  v8f o00 = {}, o01 = {}, ol0 = {};
  v8f o10 = {}, o11 = {}, ol1 = {};
#pragma unroll
  for (int r = 0; r < 8; ++r) { mr0[r] = -1e30f; mr1[r] = -1e30f; }
  const float scale = 0.17677669529663687f;  // 32^-0.5

#pragma unroll 1
  for (int mb = 0; mb < 128; ++mb) {
    int m0 = mb * 32;
    v16h bk0 = ld16h(kT + (m0 + row) * CDIM + h * HDIM + koff);
    v16h bk1 = ld16h(kT + (m0 + 16 + row) * CDIM + h * HDIM + koff);
    v8f z = {};
    v8f s00 = wmma_f16(aq0, bk0, z);
    v8f s01 = wmma_f16(aq0, bk1, z);
    v8f s10 = wmma_f16(aq1, bk0, z);
    v8f s11 = wmma_f16(aq1, bk1, z);

    int mc0 = m0 + row, mc1 = m0 + 16 + row;
    int ym0 = mc0 >> 6, xm0 = mc0 & 63;
    int ym1 = mc1 >> 6, xm1 = mc1 & 63;

#pragma unroll
    for (int qt = 0; qt < 2; ++qt) {
      v8f& sa = qt ? s10 : s00;
      v8f& sb = qt ? s11 : s01;
      float* mr = qt ? mr1 : mr0;
      v8f& oa = qt ? o10 : o00;
      v8f& ob = qt ? o11 : o01;
      v8f& ol = qt ? ol1 : ol0;
      float rmax[8];
#pragma unroll
      for (int r = 0; r < 8; ++r) {
        int nn = n0 + qt * 16 + r + hi8;
        int yn = nn >> 6, xn = nn & 63;
        int d0 = yn - ym0; d0 = d0 < 0 ? -d0 : d0;
        int e0 = xn - xm0; e0 = e0 < 0 ? -e0 : e0;
        int d1 = yn - ym1; d1 = d1 < 0 ? -d1 : d1;
        int e1 = xn - xm1; e1 = e1 < 0 ? -e1 : e1;
        sa[r] = sa[r] * scale + __expf((float)(d0 + e0) * c1);
        sb[r] = sb[r] * scale + __expf((float)(d1 + e1) * c1);
        float t = fmaxf(sa[r], sb[r]);
        t = fmaxf(t, __shfl_xor(t, 1, 32));
        t = fmaxf(t, __shfl_xor(t, 2, 32));
        t = fmaxf(t, __shfl_xor(t, 4, 32));
        t = fmaxf(t, __shfl_xor(t, 8, 32));
        rmax[r] = t;
      }
#pragma unroll
      for (int r = 0; r < 8; ++r) {
        float mn = fmaxf(mr[r], rmax[r]);
        float alpha = __expf(mr[r] - mn);
        mr[r] = mn;
        float p0 = __expf(sa[r] - mn);
        float p1 = __expf(sb[r] - mn);
        pbuf[qt * 512 + (r + hi8) * 32 + row]      = (_Float16)p0;
        pbuf[qt * 512 + (r + hi8) * 32 + 16 + row] = (_Float16)p1;
        oa[r] *= alpha;
        ob[r] *= alpha;
        ol[r] *= alpha;
      }
    }
    __syncthreads();  // single-wave WG: orders pbuf writes -> reads
    v16h pa0 = ld16h(pbuf + row * 32 + koff);
    v16h pa1 = ld16h(pbuf + 512 + row * 32 + koff);
    v16h bv0 = ld16h(vm + (h * HDIM + row) * NPIX + m0 + koff);
    v16h bv1 = ld16h(vm + (h * HDIM + 16 + row) * NPIX + m0 + koff);
    o00 = wmma_f16(pa0, bv0, o00);
    o01 = wmma_f16(pa0, bv1, o01);
    ol0 = wmma_f16(pa0, vone, ol0);
    o10 = wmma_f16(pa1, bv0, o10);
    o11 = wmma_f16(pa1, bv1, o11);
    ol1 = wmma_f16(pa1, vone, ol1);
    __syncthreads();  // WAR guard on pbuf
  }

  v8h a0, b0, a1, b1;
#pragma unroll
  for (int r = 0; r < 8; ++r) {
    a0[r] = (_Float16)(o00[r] / ol0[r]);
    b0[r] = (_Float16)(o01[r] / ol0[r]);
    a1[r] = (_Float16)(o10[r] / ol1[r]);
    b1[r] = (_Float16)(o11[r] / ol1[r]);
  }
  *(v8h*)(attn_h + (h * HDIM + row) * NPIX + n0 + hi8) = a0;
  *(v8h*)(attn_h + (h * HDIM + 16 + row) * NPIX + n0 + hi8) = b0;
  *(v8h*)(attn_h + (h * HDIM + row) * NPIX + n0 + 16 + hi8) = a1;
  *(v8h*)(attn_h + (h * HDIM + 16 + row) * NPIX + n0 + 16 + hi8) = b1;
}

// ---------------------------------------------------------------------------
static inline size_t align_up(size_t v, size_t a) { return (v + a - 1) & ~(a - 1); }

extern "C" void kernel_launch(void* const* d_in, const int* in_sizes, int n_in,
                              void* d_out, int out_size, void* d_ws, size_t ws_size,
                              hipStream_t stream) {
  (void)in_sizes; (void)n_in; (void)out_size; (void)ws_size;
  const float* x    = (const float*)d_in[0];
  const float* ow   = (const float*)d_in[1];
  const float* w3   = (const float*)d_in[2];
  const float* w5   = (const float*)d_in[3];
  const float* w7   = (const float*)d_in[4];
  const float* wop  = (const float*)d_in[5];
  const float* wq   = (const float*)d_in[6];
  const float* wk   = (const float*)d_in[7];
  const float* wv   = (const float*)d_in[8];
  const float* wo   = (const float*)d_in[9];
  const float* beta = (const float*)d_in[10];

  char* ws = (char*)d_ws;
  size_t off = 0;
  auto carve = [&](size_t bytes) { void* p = ws + off; off = align_up(off + bytes, 256); return p; };
  float*    feat  = (float*)   carve((size_t)CDIM * NPIX * 4);   // 4 MB
  float*    gxy   = (float*)   carve((size_t)2 * NPIX * 4);
  float*    wcomb = (float*)   carve((size_t)CDIM * 49 * 4);
  _Float16* xhT   = (_Float16*)carve((size_t)CDIM * NPIX * 2);   // (n,c)
  _Float16* xsT   = (_Float16*)carve((size_t)CDIM * NPIX * 2);   // (n,c)
  _Float16* wqh   = (_Float16*)carve((size_t)CDIM * CDIM * 2);
  _Float16* wkh   = (_Float16*)carve((size_t)CDIM * CDIM * 2);
  _Float16* wvh   = (_Float16*)carve((size_t)CDIM * CDIM * 2);
  _Float16* woh   = (_Float16*)carve((size_t)CDIM * CDIM * 2);
  _Float16* qT    = (_Float16*)carve((size_t)NPIX * CDIM * 2);   // (n,c)
  _Float16* kT    = (_Float16*)carve((size_t)NPIX * CDIM * 2);   // (n,c)
  _Float16* vmat  = (_Float16*)carve((size_t)CDIM * NPIX * 2);   // (c,n)
  _Float16* attnh = (_Float16*)carve((size_t)CDIM * NPIX * 2);   // (c,n)

  // 1) combined depthwise weights + f16 casts
  build_wcomb_kernel<<<(CDIM * 49 + 255) / 256, 256, 0, stream>>>(w3, w5, w7, ow, wcomb);
  cast_f16_T_kernel<<<NPIX, 256, 0, stream>>>(x, xhT);
  cast_f16_kernel<<<(CDIM * CDIM + 255) / 256, 256, 0, stream>>>(wq, wqh, CDIM * CDIM);
  cast_f16_kernel<<<(CDIM * CDIM + 255) / 256, 256, 0, stream>>>(wk, wkh, CDIM * CDIM);
  cast_f16_kernel<<<(CDIM * CDIM + 255) / 256, 256, 0, stream>>>(wv, wvh, CDIM * CDIM);
  cast_f16_kernel<<<(CDIM * CDIM + 255) / 256, 256, 0, stream>>>(wo, woh, CDIM * CDIM);

  // 2) offset branch: fused dwconv -> channel reduce -> grid sample
  dwconv_kernel<<<NPIX, 256, 0, stream>>>(x, wcomb, feat);
  offset_kernel<<<NPIX, 256, 0, stream>>>(feat, wop, gxy);
  gridsample_kernel<<<NPIX, 256, 0, stream>>>(x, gxy, xsT);

  // 3) Q/K/V projections (WMMA GEMMs, contiguous B loads from (n,c) inputs).
  dim3 ggrid(16, 32);
  gemm256_kernel<1, 1><<<ggrid, 256, 0, stream>>>(wqh, xhT, (void*)qT);
  gemm256_kernel<1, 1><<<ggrid, 256, 0, stream>>>(wkh, xsT, (void*)kT);
  gemm256_kernel<0, 1><<<ggrid, 256, 0, stream>>>(wvh, xsT, (void*)vmat);

  // 4) flash attention with decay bias (one wave per 32 rows per head)
  attention_kernel<<<dim3(NPIX / 32, NHEAD), 32, 0, stream>>>(qT, kT, vmat, beta, attnh);

  // 5) output projection -> fp32 d_out in (C,H,W) layout
  gemm256_kernel<2, 0><<<ggrid, 256, 0, stream>>>(woh, attnh, d_out);
}
